// CausalSimulationModule_70463233458698
// MI455X (gfx1250) — matrix operand order repaired and checked by reading, not compiled, not run on hardware
//
#include <hip/hip_runtime.h>
#include <math.h>

// Problem constants (B=2, N=1024, D=64 from reference)
#define B_ 2
#define N_ 1024
#define D_ 64
#define TAU_ 0.5f
#define NEG_INF_ (-1e9f)
#define EPS_ 1e-9f

typedef __attribute__((ext_vector_type(2)))  float    v2f;
typedef __attribute__((ext_vector_type(8)))  float    v8f;
typedef __attribute__((ext_vector_type(16))) _Float16 v16h;

#if __has_builtin(__builtin_amdgcn_wmma_f32_16x16x4_f32)
#define WMMA_F32_NATIVE 1
#endif

// Computes a 16x16 f32 tile of  X[16x64] @ W[64x64][:, ncol0:ncol0+16]  + acc.
// X is row-major with row stride `xstride` (LDS or global), W row-major stride `wstride`.
// Must be executed by one full wave (EXEC all ones).
__device__ __forceinline__ v8f wmma_tile_16x16x64(
    const float* __restrict__ X, int xstride,
    const float* __restrict__ W, int wstride, int ncol0,
    v8f acc, int lane)
{
  const int m  = lane & 15;   // A-frag row
  const int kh = lane >> 4;   // K half-select
  const int n  = lane & 15;   // B-frag col
#if defined(WMMA_F32_NATIVE)
  // V_WMMA_F32_16X16X4_F32: A 16x4 (2 VGPR: K = 2*kh + v), B 4x16 (same K map),
  // full f32 precision, 16 k-steps cover K=64.
#pragma unroll
  for (int k0 = 0; k0 < 64; k0 += 4) {
    const int ka = k0 + 2 * kh;
    v2f a, b;
    a.x = X[m * xstride + ka + 0];
    a.y = X[m * xstride + ka + 1];
    b.x = W[(ka + 0) * wstride + ncol0 + n];
    b.y = W[(ka + 1) * wstride + ncol0 + n];
    acc = __builtin_amdgcn_wmma_f32_16x16x4_f32(false, a, false, b,
                                                (short)0, acc, false, false);
  }
#else
  // Fallback: V_WMMA_F32_16X16X32_F16 (probe-confirmed signature), 2 k-steps.
#pragma unroll
  for (int k0 = 0; k0 < 64; k0 += 32) {
    v16h a, b;
#pragma unroll
    for (int h = 0; h < 16; ++h) {
      const int ka = k0 + (h & 7) + 8 * kh + ((h & 8) ? 16 : 0);
      const int kb = k0 + (h & 7) + 8 * kh + ((h & 8) ? 16 : 0);
      a[h] = (_Float16)X[m * xstride + ka];
      b[h] = (_Float16)W[kb * wstride + ncol0 + n];
    }
    acc = __builtin_amdgcn_wmma_f32_16x16x32_f16(false, a, false, b,
                                                 (short)0, acc, false, false);
  }
#endif
  return acc;
}

// ---------------------------------------------------------------------------
// Kernel 1: A = softmax((mask_diag(edge_logits) + gumbel(u))/tau, axis=-1)
// One block per target row t; 256 threads x 4 elements, register-resident.
// ---------------------------------------------------------------------------
__global__ void gumbel_softmax_kernel(const float* __restrict__ edge_logits,
                                      const float* __restrict__ u_noise,
                                      float* __restrict__ A)
{
  __shared__ float red[256];
  const int t   = blockIdx.x;
  const int tid = threadIdx.x;

  float z[4];
  float lmax = -INFINITY;
#pragma unroll
  for (int j = 0; j < 4; ++j) {
    const int s = tid + j * 256;
    const float logit = (s == t) ? NEG_INF_ : edge_logits[t * N_ + s];
    const float u = u_noise[t * N_ + s];
    const float g = -logf(-logf(u + EPS_) + EPS_);
    z[j] = (logit + g) * (1.0f / TAU_);
    lmax = fmaxf(lmax, z[j]);
  }
  red[tid] = lmax;
  __syncthreads();
  for (int off = 128; off > 0; off >>= 1) {
    if (tid < off) red[tid] = fmaxf(red[tid], red[tid + off]);
    __syncthreads();
  }
  lmax = red[0];
  __syncthreads();

  float lsum = 0.f;
#pragma unroll
  for (int j = 0; j < 4; ++j) {
    z[j] = expf(z[j] - lmax);
    lsum += z[j];
  }
  red[tid] = lsum;
  __syncthreads();
  for (int off = 128; off > 0; off >>= 1) {
    if (tid < off) red[tid] += red[tid + off];
    __syncthreads();
  }
  const float inv = 1.0f / red[0];
#pragma unroll
  for (int j = 0; j < 4; ++j)
    A[t * N_ + tid + j * 256] = z[j] * inv;
}

// ---------------------------------------------------------------------------
// Kernel 2 (WMMA): proj = ES@Wp + bp ; ps = proj@W1[:64] ; ptb = proj@W1[64:] + b1
// 2 waves/block, each wave owns one 16-row tile of the flattened [B*N, D] matrix.
// proj staged in LDS (stride 68 -> conflict-free A-frag reads, 16B alignment).
// ---------------------------------------------------------------------------
#define LSTR 68

__global__ void proj_ps_pt_kernel(const float* __restrict__ ES,
                                  const float* __restrict__ Wp,
                                  const float* __restrict__ bp,
                                  const float* __restrict__ W1,
                                  const float* __restrict__ b1,
                                  float* __restrict__ ps,
                                  float* __restrict__ ptb)
{
  __shared__ float stage[2 * 16 * LSTR];
  const int lane = threadIdx.x & 31;
  const int wid  = threadIdx.x >> 5;
  const int r0   = (blockIdx.x * 2 + wid) * 16;   // flattened row base
  float* proj = &stage[wid * 16 * LSTR];

  const int kh = lane >> 4, n = lane & 15;

  // proj tile -> LDS
#pragma unroll
  for (int nt = 0; nt < 4; ++nt) {
    v8f acc = {};
    acc = wmma_tile_16x16x64(ES + r0 * 64, 64, Wp, 64, nt * 16, acc, lane);
    const float bias = bp[nt * 16 + n];
#pragma unroll
    for (int r = 0; r < 8; ++r)
      proj[(r + 8 * kh) * LSTR + nt * 16 + n] = acc[r] + bias;
  }
  __syncthreads();

  // ps / ptb tiles -> workspace
#pragma unroll
  for (int nt = 0; nt < 4; ++nt) {
    v8f accs = {}, acct = {};
    accs = wmma_tile_16x16x64(proj, LSTR, W1,           64, nt * 16, accs, lane);
    acct = wmma_tile_16x16x64(proj, LSTR, W1 + 64 * 64, 64, nt * 16, acct, lane);
    const float bias1 = b1[nt * 16 + n];
#pragma unroll
    for (int r = 0; r < 8; ++r) {
      const int row = r0 + r + 8 * kh;
      const int col = nt * 16 + n;
      ps [row * 64 + col] = accs[r];
      ptb[row * 64 + col] = acct[r] + bias1;   // fold b1 into target term
    }
  }
}

// ---------------------------------------------------------------------------
// Kernel 3 (dominant): hbar[b,t,d] = sum_s A[t,s]*relu(ps[b,s,d]+ptb[b,t,d]),
// then next_state tile = ES + hbar@W2 + b2 (b2 exact: rows of A sum to 1).
// 256 threads = (16 targets) x (16 dim-quads); float4 loads -> 4x fewer vmem
// instructions and 4 independent FMA chains per thread. Wave 0 runs the WMMA
// epilogue on the LDS-staged hbar tile.
// ---------------------------------------------------------------------------
__global__ void __launch_bounds__(256)
aggregate_kernel(const float* __restrict__ ES,
                 const float* __restrict__ A,
                 const float* __restrict__ ps,
                 const float* __restrict__ ptb,
                 const float* __restrict__ W2,
                 const float* __restrict__ b2,
                 float* __restrict__ out)
{
  __shared__ float hbar[16 * LSTR];
  const int base = blockIdx.x * 16;     // flattened row base (never straddles batch)
  const int b    = base / N_;
  const int t0   = base % N_;
  const int tl   = threadIdx.x >> 4;    // 0..15 target within tile
  const int dq   = threadIdx.x & 15;    // dim quad: covers d = 4*dq .. 4*dq+3

  const float4 ptv = *(const float4*)&ptb[(base + tl) * 64 + 4 * dq];
  const float* __restrict__ Arow = A + (size_t)(t0 + tl) * N_;
  const float4* __restrict__ psq =
      (const float4*)(ps + (size_t)b * N_ * 64) + dq;   // stride 16 float4 per s

  float4 acc = make_float4(0.f, 0.f, 0.f, 0.f);
#pragma unroll 4
  for (int s = 0; s < N_; ++s) {
    const float a = Arow[s];
    const float4 x = psq[s * 16];
    acc.x = fmaf(a, fmaxf(x.x + ptv.x, 0.f), acc.x);
    acc.y = fmaf(a, fmaxf(x.y + ptv.y, 0.f), acc.y);
    acc.z = fmaf(a, fmaxf(x.z + ptv.z, 0.f), acc.z);
    acc.w = fmaf(a, fmaxf(x.w + ptv.w, 0.f), acc.w);
  }
  *(float4*)&hbar[tl * LSTR + 4 * dq] = acc;   // 16B-aligned (LSTR*4 % 16 == 0)
  __syncthreads();

  if (threadIdx.x < 32) {               // one full wave: EXEC all ones for WMMA
    const int lane = threadIdx.x;
    const int kh = lane >> 4, n = lane & 15;
#pragma unroll
    for (int nt = 0; nt < 4; ++nt) {
      v8f accv = {};
      accv = wmma_tile_16x16x64(hbar, LSTR, W2, 64, nt * 16, accv, lane);
      const float bias2 = b2[nt * 16 + n];
#pragma unroll
      for (int r = 0; r < 8; ++r) {
        const int row = base + r + 8 * kh;
        const int col = nt * 16 + n;
        out[row * 64 + col] = ES[row * 64 + col] + accv[r] + bias2;
      }
    }
  }
}

// ---------------------------------------------------------------------------
extern "C" void kernel_launch(void* const* d_in, const int* in_sizes, int n_in,
                              void* d_out, int out_size, void* d_ws, size_t ws_size,
                              hipStream_t stream) {
  const float* ES = (const float*)d_in[0];   // entity_states [B,N,D]
  const float* U  = (const float*)d_in[1];   // u_noise [N,N]
  const float* Wp = (const float*)d_in[2];   // W_proj [D,D]
  const float* bp = (const float*)d_in[3];   // b_proj [D]
  const float* EL = (const float*)d_in[4];   // edge_logits [N,N]
  const float* W1 = (const float*)d_in[5];   // W1 [2D,D]
  const float* b1 = (const float*)d_in[6];   // b1 [D]
  const float* W2 = (const float*)d_in[7];   // W2 [D,D]
  const float* b2 = (const float*)d_in[8];   // b2 [D]

  float* out  = (float*)d_out;               // next_state [B,N,D]
  float* Aout = out + (size_t)B_ * N_ * D_;  // A [N,N] (2nd tuple output)

  float* ps  = (float*)d_ws;                 // [B*N, D]
  float* ptb = ps + (size_t)B_ * N_ * D_;    // [B*N, D] (pt + b1)

  gumbel_softmax_kernel<<<dim3(N_), dim3(256), 0, stream>>>(EL, U, Aout);
  proj_ps_pt_kernel<<<dim3(B_ * N_ / 32), dim3(64), 0, stream>>>(ES, Wp, bp, W1, b1, ps, ptb);
  aggregate_kernel<<<dim3(B_ * N_ / 16), dim3(256), 0, stream>>>(ES, Aout, ps, ptb, W2, b2, out);
}
